// MultiHeadAttentionLayerGritSparse_84954453115473
// MI455X (gfx1250) — matrix operand order repaired and checked by reading, not compiled, not run on hardware
//
#include <hip/hip_runtime.h>
#include <hip/hip_bf16.h>
#include <math.h>

typedef __attribute__((ext_vector_type(16))) _Float16 v16h;
typedef __attribute__((ext_vector_type(8)))  float    v8f;

#define NN      50000
#define NE      800000
#define INDIM   64
#define HID     64      // HEADS * OUT_DIM
#define HEADS_  8
#define ODIM    8

// ---------------------------------------------------------------------------
// A-fragment (16x32 f16, ISA 7.12.2): lane L -> row m0+(L&15);
//   halves 0..7  -> K = kb + 8*(L>>4) + h
//   halves 8..15 -> K = kb + 16 + 8*(L>>4) + (h-8)
// fp32 split into hi+lo f16 for ~fp32 accuracy (3 WMMAs per product).
// Vectorized: two 32-byte runs -> 4x global_load_b128.
// ---------------------------------------------------------------------------
__device__ __forceinline__ void load_a_split(const float* __restrict__ src, int ld,
                                             int m0, int kb, int lane,
                                             v16h& hi, v16h& lo) {
    int row = m0 + (lane & 15);
    int grp = lane >> 4;
    const float* p = src + (size_t)row * ld + kb + grp * 8;   // 32B aligned
    float f[16];
    *(float4*)&f[0]  = *(const float4*)(p);
    *(float4*)&f[4]  = *(const float4*)(p + 4);
    *(float4*)&f[8]  = *(const float4*)(p + 16);
    *(float4*)&f[12] = *(const float4*)(p + 20);
#pragma unroll
    for (int h = 0; h < 16; ++h) {
        _Float16 hh = (_Float16)f[h];
        hi[h] = hh;
        lo[h] = (_Float16)(f[h] - (float)hh);
    }
}

// Packed B fragments: frag = kbIdx*NT + nt; per lane 32 contiguous halves
// (16 hi then 16 lo) -> 64B aligned, 4x b128 loads, zero conversion VALU.
__device__ __forceinline__ void load_b_packed(const _Float16* __restrict__ pk,
                                              int frag, int lane,
                                              v16h& hi, v16h& lo) {
    const _Float16* p = pk + ((size_t)frag * 32 + lane) * 32;
    hi = *(const v16h*)(p);
    lo = *(const v16h*)(p + 16);
}

// C += A*B with f16-split emulation (drop lo*lo term)
__device__ __forceinline__ v8f wmma3(v16h ah, v16h al, v16h bh, v16h bl, v8f c) {
    c = __builtin_amdgcn_wmma_f32_16x16x32_f16(false, ah, false, bh, (short)0, c, false, false);
    c = __builtin_amdgcn_wmma_f32_16x16x32_f16(false, ah, false, bl, (short)0, c, false, false);
    c = __builtin_amdgcn_wmma_f32_16x16x32_f16(false, al, false, bh, (short)0, c, false, false);
    return c;
}

// ---------------------------------------------------------------------------
// Kernel P: pack one weight matrix (Kdim=64 x N) into B-fragment layout.
// grid = 2 * (N/16) blocks of 1 wave; block b = kbIdx*(N/16) + nt.
// B (32x16 f16): lane L -> col = n0+(L&15); half h<8 -> K = kb + 8*(L>>4) + h, h>=8 -> K = kb + 16 + 8*(L>>4) + (h-8)
// ---------------------------------------------------------------------------
__global__ void __launch_bounds__(32) pack_w_kernel(const float* __restrict__ W,
                                                    int N, _Float16* __restrict__ out) {
    int lane = threadIdx.x & 31;
    int frag = blockIdx.x;
    int NT   = N >> 4;
    int kb   = (frag / NT) * 32;
    int nt   = frag % NT;
    int col  = nt * 16 + (lane & 15);
    int grp  = lane >> 4;
    _Float16* o = out + ((size_t)frag * 32 + lane) * 32;
#pragma unroll
    for (int h = 0; h < 16; ++h) {
        float f = W[(size_t)(kb + grp * 8 + h + (h >= 8 ? 8 : 0)) * N + col];
        _Float16 hh = (_Float16)f;
        o[h]      = hh;
        o[16 + h] = (_Float16)(f - (float)hh);
    }
}

// ---------------------------------------------------------------------------
// Kernel 0: zero the atomic accumulators
// ---------------------------------------------------------------------------
__global__ void zero_kernel(float* __restrict__ p, long n) {
    long i = (long)blockIdx.x * blockDim.x + threadIdx.x;
    if (i < n) p[i] = 0.0f;
}

// ---------------------------------------------------------------------------
// Kernel 1: node projections Q/K/V = x @ W + b (one wave per 16-node tile)
// ---------------------------------------------------------------------------
__device__ __forceinline__ void proj_tile(const _Float16* __restrict__ pkW,
                                          const float* __restrict__ b,
                                          float* __restrict__ O,
                                          int m0, int lane,
                                          v16h a0h, v16h a0l, v16h a1h, v16h a1l) {
#pragma unroll
    for (int nt = 0; nt < 4; ++nt) {
        v8f acc = {};
        v16h bh, bl;
        load_b_packed(pkW, nt, lane, bh, bl);          // kb = 0
        acc = wmma3(a0h, a0l, bh, bl, acc);
        load_b_packed(pkW, 4 + nt, lane, bh, bl);      // kb = 32
        acc = wmma3(a1h, a1l, bh, bl, acc);
        int col  = nt * 16 + (lane & 15);
        float bb = b[col];
        int mb   = m0 + ((lane >> 4) << 3);
#pragma unroll
        for (int r = 0; r < 8; ++r)
            O[(size_t)(mb + r) * HID + col] = acc[r] + bb;   // C: M=r+8*grp, N=col
    }
}

__global__ void __launch_bounds__(32) qkv_kernel(const float* __restrict__ x,
                                                 const _Float16* __restrict__ pkWq, const float* __restrict__ bq,
                                                 const _Float16* __restrict__ pkWk, const float* __restrict__ bk,
                                                 const _Float16* __restrict__ pkWv, const float* __restrict__ bv,
                                                 float* __restrict__ Q,
                                                 float* __restrict__ K,
                                                 float* __restrict__ V) {
    int lane = threadIdx.x & 31;
    int m0   = blockIdx.x * 16;      // NN = 3125 * 16
    v16h a0h, a0l, a1h, a1l;
    load_a_split(x, INDIM, m0, 0,  lane, a0h, a0l);
    load_a_split(x, INDIM, m0, 32, lane, a1h, a1l);
    proj_tile(pkWq, bq, Q, m0, lane, a0h, a0l, a1h, a1l);
    proj_tile(pkWk, bk, K, m0, lane, a0h, a0l, a1h, a1l);
    proj_tile(pkWv, bv, V, m0, lane, a0h, a0l, a1h, a1l);
}

// ---------------------------------------------------------------------------
// Kernel 2: fused edge GEMM (E tile in LDS, never hits HBM) + scoring +
//           clamped-exp softmax numerators via atomics. One wave / 16 edges.
// ---------------------------------------------------------------------------
__global__ void __launch_bounds__(32) edge_kernel(const float* __restrict__ edge_attr,
                                                  const int*   __restrict__ edge_index,
                                                  const _Float16* __restrict__ pkWe,
                                                  const float* __restrict__ be,
                                                  const float* __restrict__ Aw,
                                                  const float* __restrict__ Q,
                                                  const float* __restrict__ K,
                                                  const float* __restrict__ V,
                                                  float* __restrict__ sumbuf,
                                                  float* __restrict__ wVnum,
                                                  float* __restrict__ rowVnum,
                                                  float* __restrict__ wE_out) {
    __shared__ float etile[16 * 128];     // 8 KB: 16 edges x (H*2D)
    int lane = threadIdx.x & 31;
    int m0   = blockIdx.x * 16;           // NE = 50000 * 16

    // --- GEMM phase ---
    v16h a0h, a0l, a1h, a1l;
    load_a_split(edge_attr, INDIM, m0, 0,  lane, a0h, a0l);
    load_a_split(edge_attr, INDIM, m0, 32, lane, a1h, a1l);
#pragma unroll
    for (int nt = 0; nt < 8; ++nt) {
        v8f acc = {};
        v16h bh, bl;
        load_b_packed(pkWe, nt, lane, bh, bl);         // kb = 0
        acc = wmma3(a0h, a0l, bh, bl, acc);
        load_b_packed(pkWe, 8 + nt, lane, bh, bl);     // kb = 32
        acc = wmma3(a1h, a1l, bh, bl, acc);
        int col  = nt * 16 + (lane & 15);
        float bb = be[col];
        int mb   = (lane >> 4) << 3;
#pragma unroll
        for (int r = 0; r < 8; ++r)
            etile[(mb + r) * 128 + col] = acc[r] + bb;
    }
    __syncthreads();

    // --- attention phase: 16 edges x 8 heads = 128 pairs over 32 lanes ---
#pragma unroll
    for (int it = 0; it < 4; ++it) {
        int pair = it * 32 + lane;
        int el   = pair >> 3;             // edge in tile
        int h    = pair & 7;              // head
        int eg   = m0 + el;
        int s    = edge_index[eg];        // src
        int d    = edge_index[NE + eg];   // dst
        const float* Kp = K + (size_t)s * HID + h * ODIM;
        const float* Qp = Q + (size_t)d * HID + h * ODIM;
        const float* Vp = V + (size_t)s * HID + h * ODIM;
        const float* ew = &etile[el * 128 + h * 16];   // E_w[0..7], E_b[8..15]

        float kq[8], vv[8], eww[8], ebb[8];
        *(float4*)&kq[0]  = *(const float4*)(Kp);
        *(float4*)&kq[4]  = *(const float4*)(Kp + 4);
        {
            float4 q0 = *(const float4*)(Qp);
            float4 q1 = *(const float4*)(Qp + 4);
            kq[0] += q0.x; kq[1] += q0.y; kq[2] += q0.z; kq[3] += q0.w;
            kq[4] += q1.x; kq[5] += q1.y; kq[6] += q1.z; kq[7] += q1.w;
        }
        *(float4*)&vv[0]  = *(const float4*)(Vp);
        *(float4*)&vv[4]  = *(const float4*)(Vp + 4);
        *(float4*)&eww[0] = *(const float4*)(ew);
        *(float4*)&eww[4] = *(const float4*)(ew + 4);
        *(float4*)&ebb[0] = *(const float4*)(ew + 8);
        *(float4*)&ebb[4] = *(const float4*)(ew + 12);

        float et[8];
        float a = 0.0f;
#pragma unroll
        for (int c = 0; c < 8; ++c) {
            float t  = kq[c] * eww[c];
            float ss = copysignf(sqrtf(fabsf(t)), t) + ebb[c];   // signed_sqrt + E_b
            float e  = fmaxf(ss, 0.0f);                           // relu
            et[c] = e;
            a += e * Aw[c * HEADS_ + h];                          // Aw[d,h,0]
        }
        a = fminf(fmaxf(a, -5.0f), 5.0f);
        float p = __expf(a);   // clamp => exp stable, no segment-max pass needed

        // wE output = e_t (coalesced-ish contiguous 32B per pair)
        float* wp = wE_out + (size_t)eg * HID + h * ODIM;
        *(float4*)(wp)     = *(const float4*)&et[0];
        *(float4*)(wp + 4) = *(const float4*)&et[4];

        // softmax numerators (denominator applied in finalize)
        atomicAdd(&sumbuf[(size_t)d * HEADS_ + h], p);
        float* wv = wVnum   + ((size_t)d * HEADS_ + h) * ODIM;
        float* rv = rowVnum + ((size_t)d * HEADS_ + h) * ODIM;
#pragma unroll
        for (int c = 0; c < 8; ++c) {
            atomicAdd(&wv[c], vv[c] * p);
            atomicAdd(&rv[c], et[c] * p);
        }
    }
}

// ---------------------------------------------------------------------------
// Kernel 3: per-(node,head) finalize: divide by denom, apply VeRow, write h_out
// ---------------------------------------------------------------------------
__global__ void finalize_kernel(const float* __restrict__ sumbuf,
                                const float* __restrict__ wVnum,
                                const float* __restrict__ rowVnum,
                                const float* __restrict__ VeRow,
                                float* __restrict__ h_out) {
    int i = blockIdx.x * blockDim.x + threadIdx.x;   // (node, head)
    if (i >= NN * HEADS_) return;
    int n = i >> 3;
    int h = i & 7;
    float inv = 1.0f / (sumbuf[i] + 1e-16f);
    float rv[8];
#pragma unroll
    for (int dd = 0; dd < 8; ++dd) rv[dd] = rowVnum[(size_t)i * ODIM + dd] * inv;
#pragma unroll
    for (int c = 0; c < 8; ++c) {
        float acc = wVnum[(size_t)i * ODIM + c] * inv;
#pragma unroll
        for (int dd = 0; dd < 8; ++dd)
            acc += rv[dd] * VeRow[dd * (HEADS_ * ODIM) + h * ODIM + c];  // VeRow[d,h,c]
        h_out[(size_t)n * HID + h * ODIM + c] = acc;
    }
}

// ---------------------------------------------------------------------------
// Host launcher
// ---------------------------------------------------------------------------
extern "C" void kernel_launch(void* const* d_in, const int* in_sizes, int n_in,
                              void* d_out, int out_size, void* d_ws, size_t ws_size,
                              hipStream_t stream) {
    const float* x         = (const float*)d_in[0];
    const float* edge_attr = (const float*)d_in[1];
    const int*   edge_idx  = (const int*)  d_in[2];
    const float* Wq        = (const float*)d_in[3];
    const float* bq        = (const float*)d_in[4];
    const float* Wk        = (const float*)d_in[5];
    const float* bk        = (const float*)d_in[6];
    const float* We        = (const float*)d_in[7];
    const float* be        = (const float*)d_in[8];
    const float* Wv        = (const float*)d_in[9];
    const float* bv        = (const float*)d_in[10];
    const float* Aw        = (const float*)d_in[11];
    const float* VeRow     = (const float*)d_in[12];

    float* ws = (float*)d_ws;
    float* Q       = ws;                          // 3,200,000 floats
    float* K       = ws + 3200000;
    float* V       = ws + 6400000;
    float* sumbuf  = ws + 9600000;                //   400,000
    float* wVnum   = ws + 10000000;               // 3,200,000
    float* rowVnum = ws + 13200000;               // 3,200,000
    // packed f16 hi/lo weight fragments (64B-aligned region after fp32 scratch)
    _Float16* pkWq = (_Float16*)(ws + 16400000);  // 8 frags * 1024 halves
    _Float16* pkWk = pkWq + 8 * 1024;
    _Float16* pkWv = pkWk + 8 * 1024;
    _Float16* pkWe = pkWv + 8 * 1024;             // 16 frags * 1024 halves

    float* h_out = (float*)d_out;                        // NN*64
    float* wE    = (float*)d_out + (size_t)NN * HID;     // NE*64

    // P) pack weights into WMMA B-fragment layout (once per launch, tiny)
    pack_w_kernel<<< 8, 32, 0, stream>>>(Wq, HID, pkWq);
    pack_w_kernel<<< 8, 32, 0, stream>>>(Wk, HID, pkWk);
    pack_w_kernel<<< 8, 32, 0, stream>>>(Wv, HID, pkWv);
    pack_w_kernel<<<16, 32, 0, stream>>>(We, 128, pkWe);

    // 0) zero atomic accumulators (sumbuf + wVnum + rowVnum = 6.8M floats)
    {
        long n = 6800000;
        zero_kernel<<<(int)((n + 255) / 256), 256, 0, stream>>>(sumbuf, n);
    }
    // 1) node projections (WMMA)
    qkv_kernel<<<NN / 16, 32, 0, stream>>>(x, pkWq, bq, pkWk, bk, pkWv, bv, Q, K, V);
    // 2) fused edge GEMM + attention (WMMA + atomics)
    edge_kernel<<<NE / 16, 32, 0, stream>>>(edge_attr, edge_idx, pkWe, be, Aw,
                                            Q, K, V, sumbuf, wVnum, rowVnum, wE);
    // 3) finalize per (node, head)
    {
        int n = NN * HEADS_;
        finalize_kernel<<<(n + 255) / 256, 256, 0, stream>>>(sumbuf, wVnum, rowVnum,
                                                             VeRow, h_out);
    }
}
